// LiquidRNN_63788854280638
// MI455X (gfx1250) — matrix operand order repaired and checked by reading, not compiled, not run on hardware
//
#include <hip/hip_runtime.h>

// ---------------------------------------------------------------------------
// LiquidRNN on MI455X (gfx1250, wave32, WMMA).
//   B=64, S=1024, D=256, H=512, BU=512
// Sequential recurrence over S -> persistent kernel, 4 WGs (one per 16-row
// batch tile), 16 waves each (one wave = 2 of the 32 16-col N-tiles).
// x@Wb_x precomputed for all timesteps (no recurrent dep) when ws allows.
// ---------------------------------------------------------------------------

#define BB   64
#define SS   1024
#define DD   256
#define HH   512
#define BU   512
#define KTB  24   // (D+H)/32 k-tiles for backbone GEMM
#define KTH  16   // BU/32 k-tiles for head GEMM

typedef __attribute__((ext_vector_type(16))) __bf16 v16bf;
typedef __attribute__((ext_vector_type(8)))  float  v8f;

// ---- WMMA operand layouts (ISA 7.12.2, wave32) -----------------------------
// A 16x32 bf16: lane = half*16 + m (half=lane/16), 16 elems/lane:
//   e=0..7  -> K = half*8 + e ; e=8..15 -> K = 16 + half*8 + (e-8)
// A-layout LDS slot for element (m, k):
__device__ __forceinline__ int a_slot(int m, int k) {
    int kt = k >> 5, kk = k & 31;
    int hf = (kk >> 3) & 1;
    int e  = (kk & 7) | ((kk & 16) >> 1);
    return kt * 512 + (hf * 16 + m) * 16 + e;
}
// B 32x16 bf16: lane = half*16 + (n%16), same K order as A (e<8: half*8+e; e>=8: 16+half*8+(e-8)).
// Packed weights: [ntile][ktile][lane][e] contiguous (32B per lane per tile).

// ---------------------------------------------------------------------------
// Pack fp32 weights [K, N] row-major -> bf16 WMMA-B tile layout.
// ---------------------------------------------------------------------------
__global__ void pack_w_kernel(const float* __restrict__ W, __bf16* __restrict__ dst,
                              int K, int N, int KT) {
    int i = blockIdx.x * blockDim.x + threadIdx.x;
    if (i >= K * N) return;
    int k = i / N, n = i % N;
    int kt = k >> 5, kk = k & 31;
    int hf = (kk >> 3) & 1, e = (kk & 7) | ((kk & 16) >> 1);  // B-operand layout
    int nt = n >> 4, nn = n & 15;
    size_t slot = ((size_t)(nt * KT + kt) * 32 + (hf * 16 + nn)) * 16 + e;
    dst[slot] = (__bf16)W[(size_t)k * N + n];
}

// ---------------------------------------------------------------------------
// Zx = X @ Wb[0:D,:] + bb for all B*S rows.  Fully parallel: 4096 blocks.
// ---------------------------------------------------------------------------
__global__ __launch_bounds__(512, 1)
void zx_gemm_kernel(const float* __restrict__ x, const __bf16* __restrict__ WbP,
                    const float* __restrict__ bb, float* __restrict__ Zx) {
    __shared__ __bf16 As[8 * 512];          // 8 k-tiles of A (16x256), 8 KB
    const int tid = threadIdx.x, wave = tid >> 5, lane = tid & 31;
    const int hf = lane >> 4, l16 = lane & 15;
    const size_t row0 = (size_t)blockIdx.x * 16;

    // stage 16x256 fp32 -> bf16 A-layout (coalesced reads)
    #pragma unroll
    for (int j = 0; j < 8; ++j) {
        int i = tid + j * 512;
        int m = i >> 8, k = i & 255;
        As[a_slot(m, k)] = (__bf16)x[(row0 + m) * DD + k];
    }
    __syncthreads();

    #pragma unroll
    for (int s = 0; s < 2; ++s) {
        const int nt = wave + s * 16;
        v8f acc;
        #pragma unroll
        for (int r = 0; r < 8; ++r) acc[r] = 0.0f;
        #pragma unroll
        for (int kt = 0; kt < 8; ++kt) {
            v16bf a = *(const v16bf*)&As[kt * 512 + lane * 16];
            v16bf b = *(const v16bf*)&WbP[((size_t)(nt * KTB + kt) * 32 + lane) * 16];
            acc = __builtin_amdgcn_wmma_f32_16x16x32_bf16(false, a, false, b,
                                                          (short)0, acc, false, false);
        }
        const int n = nt * 16 + l16;
        const float bv = bb[n];
        #pragma unroll
        for (int r = 0; r < 8; ++r) {
            int m = r + hf * 8;
            Zx[(row0 + m) * BU + n] = acc[r] + bv;
        }
    }
}

// ---------------------------------------------------------------------------
// Persistent recurrent kernel: 4 blocks x 512 threads. Loops t = 0..S-1.
// ---------------------------------------------------------------------------
__global__ __launch_bounds__(512, 1)
void liquid_rnn_kernel(const float* __restrict__ x,
                       const __bf16* __restrict__ WbP, const __bf16* __restrict__ WhP,
                       const float* __restrict__ bb,  const float* __restrict__ bh,
                       const float* __restrict__ tau, const float* __restrict__ Zx,
                       float* __restrict__ out, int use_zx) {
    __shared__ __bf16 A1[KTB * 512];        // [x | h] in A-layout, 24 KB
    __shared__ __bf16 A2[KTH * 512];        // z in A-layout, 16 KB

    const int tid = threadIdx.x, wave = tid >> 5, lane = tid & 31;
    const int hf = lane >> 4, l16 = lane & 15;
    const int m0 = blockIdx.x * 16;         // batch-row tile

    // zero A1 (h region must be 0 at t=0; x region overwritten per step)
    #pragma unroll
    for (int j = 0; j < KTB; ++j) A1[tid + j * 512] = (__bf16)0.0f;

    // per-lane state: h in registers (lane owns 2 n-columns x 8 m-rows)
    float hreg[2][8];
    float bbv[2], bhv[2], itau[2];
    #pragma unroll
    for (int s = 0; s < 2; ++s) {
        int n = (wave + s * 16) * 16 + l16;
        bbv[s]  = bb[n];
        bhv[s]  = bh[n];
        itau[s] = 1.0f / tau[n];
        #pragma unroll
        for (int r = 0; r < 8; ++r) hreg[s][r] = 0.0f;
    }
    __syncthreads();

    for (int t = 0; t < SS; ++t) {
        // ---- stage x_t (only needed when Zx not precomputed) --------------
        if (!use_zx) {
            #pragma unroll
            for (int j = 0; j < 8; ++j) {
                int i = tid + j * 512;
                int m = i >> 8, k = i & 255;
                A1[a_slot(m, k)] = (__bf16)x[((size_t)(m0 + m) * SS + t) * DD + k];
            }
        }
        __syncthreads();

        // ---- GEMM1: z = [x,h] @ Wb (+bb) ---------------------------------
        #pragma unroll
        for (int s = 0; s < 2; ++s) {
            const int nt = wave + s * 16;
            const int n  = nt * 16 + l16;
            v8f acc;
            if (use_zx) {
                #pragma unroll
                for (int r = 0; r < 8; ++r)
                    acc[r] = Zx[((size_t)(m0 + r + hf * 8) * SS + t) * BU + n];
            } else {
                #pragma unroll
                for (int r = 0; r < 8; ++r) acc[r] = bbv[s];
                #pragma unroll
                for (int kt = 0; kt < 8; ++kt) {   // x part of K
                    v16bf a = *(const v16bf*)&A1[kt * 512 + lane * 16];
                    v16bf b = *(const v16bf*)&WbP[((size_t)(nt * KTB + kt) * 32 + lane) * 16];
                    acc = __builtin_amdgcn_wmma_f32_16x16x32_bf16(false, a, false, b,
                                                                  (short)0, acc, false, false);
                }
            }
            #pragma unroll
            for (int kt = 8; kt < KTB; ++kt) {     // h part of K (always)
                v16bf a = *(const v16bf*)&A1[kt * 512 + lane * 16];
                v16bf b = *(const v16bf*)&WbP[((size_t)(nt * KTB + kt) * 32 + lane) * 16];
                acc = __builtin_amdgcn_wmma_f32_16x16x32_bf16(false, a, false, b,
                                                              (short)0, acc, false, false);
            }
            // z -> A2 in A-layout (k index for GEMM2 is n)
            const int sbase = (n >> 5) * 512 + (((n >> 3) & 1) * 16) * 16
                            + ((n & 7) | ((n & 16) >> 1));
            #pragma unroll
            for (int r = 0; r < 8; ++r) {
                int m = r + hf * 8;
                A2[sbase + m * 16] = (__bf16)acc[r];
            }
        }
        __syncthreads();

        // ---- GEMM2 + tanh + liquid ODE update ----------------------------
        #pragma unroll
        for (int s = 0; s < 2; ++s) {
            const int nt = wave + s * 16;
            const int n  = nt * 16 + l16;
            v8f acc;
            #pragma unroll
            for (int r = 0; r < 8; ++r) acc[r] = bhv[s];
            #pragma unroll
            for (int kt = 0; kt < KTH; ++kt) {
                v16bf a = *(const v16bf*)&A2[kt * 512 + lane * 16];
                v16bf b = *(const v16bf*)&WhP[((size_t)(nt * KTH + kt) * 32 + lane) * 16];
                acc = __builtin_amdgcn_wmma_f32_16x16x32_bf16(false, a, false, b,
                                                              (short)0, acc, false, false);
            }
            const int sbase = (n >> 5) * 512 + (((n >> 3) & 1) * 16) * 16
                            + ((n & 7) | ((n & 16) >> 1)) + 8 * 512; // k = D + n
            #pragma unroll
            for (int r = 0; r < 8; ++r) {
                int m = r + hf * 8;
                float pre = acc[r];
                // tanh via exp on |x| (no NaN at +/-inf), sign fixup
                float ax = __builtin_fabsf(pre);
                float e  = __expf(-2.0f * ax);
                float th = (1.0f - e) / (1.0f + e);
                th = (pre < 0.0f) ? -th : th;
                float h  = hreg[s][r];
                float hn = h + (th - h) * itau[s];
                hreg[s][r] = hn;
                out[((size_t)(m0 + m) * SS + t) * HH + n] = hn;  // return_sequences
                A1[sbase + m * 16] = (__bf16)hn;                 // h_{t+1} for GEMM1
            }
        }
        __syncthreads();
    }
}

// ---------------------------------------------------------------------------
extern "C" void kernel_launch(void* const* d_in, const int* in_sizes, int n_in,
                              void* d_out, int out_size, void* d_ws, size_t ws_size,
                              hipStream_t stream) {
    const float* x   = (const float*)d_in[0];  // [B,S,D]
    const float* Wb  = (const float*)d_in[1];  // [D+H, BU]
    const float* bb  = (const float*)d_in[2];  // [BU]
    const float* Wh  = (const float*)d_in[3];  // [BU, H]
    const float* bh  = (const float*)d_in[4];  // [H]
    const float* tau = (const float*)d_in[5];  // [H]
    float* out = (float*)d_out;

    char* ws = (char*)d_ws;
    const size_t wb_bytes = (size_t)(DD + HH) * BU * sizeof(__bf16);  // 768 KB
    const size_t wh_bytes = (size_t)BU * HH * sizeof(__bf16);         // 512 KB
    const size_t zx_bytes = (size_t)BB * SS * BU * sizeof(float);     // 128 MB
    __bf16* WbP = (__bf16*)ws;
    __bf16* WhP = (__bf16*)(ws + wb_bytes);
    float*  Zx  = (float*)(ws + wb_bytes + wh_bytes);
    const int use_zx = (ws_size >= wb_bytes + wh_bytes + zx_bytes) ? 1 : 0;

    // 1) pack weights to bf16 WMMA-B layout (L2-resident thereafter)
    pack_w_kernel<<<((DD + HH) * BU + 255) / 256, 256, 0, stream>>>(Wb, WbP, DD + HH, BU, KTB);
    pack_w_kernel<<<(BU * HH + 255) / 256, 256, 0, stream>>>(Wh, WhP, BU, HH, KTH);

    // 2) hoist the non-recurrent x@Wb_x + bb over all timesteps (whole-chip GEMM)
    if (use_zx)
        zx_gemm_kernel<<<(BB * SS) / 16, 512, 0, stream>>>(x, WbP, bb, Zx);

    // 3) persistent recurrent kernel: one WG per 16-row batch tile
    liquid_rnn_kernel<<<BB / 16, 512, 0, stream>>>(x, WbP, WhP, bb, bh, tau, Zx, out, use_zx);
}
